// DistSparseMoe_11630771437974
// MI455X (gfx1250) — matrix operand order repaired and checked by reading, not compiled, not run on hardware
//
#include <hip/hip_runtime.h>
#include <math.h>

typedef __attribute__((ext_vector_type(2)))  float  v2f;
typedef __attribute__((ext_vector_type(4)))  float  v4f;
typedef __attribute__((ext_vector_type(8)))  float  v8f;
typedef __attribute__((ext_vector_type(8)))  __bf16 v8bf;
typedef __attribute__((ext_vector_type(16))) __bf16 v16bf;

#define NTOK 8192   // B*S
#define DIM  1024   // D
#define NEXP 8
#define CAP  1024   // (NTOK/NEXP)*CF

// ---------------- bf16 split helpers ----------------------------------------
__device__ __forceinline__ float hi_part(float x) {
    return __uint_as_float(__float_as_uint(x) & 0xFFFF0000u);
}
__device__ __forceinline__ unsigned pack_hi2(float a, float b) {
    // truncated bf16(a) in low half, bf16(b) in high half
    return (__float_as_uint(a) >> 16) | (__float_as_uint(b) & 0xFFFF0000u);
}
__device__ __forceinline__ unsigned short bf16b(float f) {
    __bf16 h = (__bf16)f;                 // round-to-nearest-even
    return __builtin_bit_cast(unsigned short, h);
}
__device__ __forceinline__ unsigned pack_lo2(float a, float b) {
    return (unsigned)bf16b(a - hi_part(a)) | ((unsigned)bf16b(b - hi_part(b)) << 16);
}

union fragU { v16bf v; v8bf h[2]; };

// ---------------- Kernel 1: router (full f32; argmax must not flip) ---------
__global__ __launch_bounds__(256) void router_kernel(const float* __restrict__ h,
                                                     const float* __restrict__ Wg,
                                                     float* __restrict__ gate,
                                                     int* __restrict__ best) {
    int wave = threadIdx.x >> 5;
    int lane = threadIdx.x & 31;
    int token = blockIdx.x * 8 + wave;
    const float* hrow = h + (size_t)token * DIM;

    float acc[NEXP];
#pragma unroll
    for (int e = 0; e < NEXP; ++e) acc[e] = 0.0f;

    for (int k = lane; k < DIM; k += 32) {
        float x = hrow[k];
        v4f w0 = *(const v4f*)(Wg + (size_t)k * NEXP);
        v4f w1 = *(const v4f*)(Wg + (size_t)k * NEXP + 4);
        acc[0] += x * w0.x; acc[1] += x * w0.y; acc[2] += x * w0.z; acc[3] += x * w0.w;
        acc[4] += x * w1.x; acc[5] += x * w1.y; acc[6] += x * w1.z; acc[7] += x * w1.w;
    }
#pragma unroll
    for (int e = 0; e < NEXP; ++e) {
#pragma unroll
        for (int off = 16; off > 0; off >>= 1)
            acc[e] += __shfl_xor(acc[e], off, 32);
    }
    if (lane == 0) {
        float m = acc[0]; int bi = 0;
#pragma unroll
        for (int e = 1; e < NEXP; ++e)
            if (acc[e] > m) { m = acc[e]; bi = e; }
        float s = 0.0f;
#pragma unroll
        for (int e = 0; e < NEXP; ++e) s += expf(acc[e] - m);
        gate[token] = 1.0f / s;   // softmax value at the argmax
        best[token] = bi;
    }
}

// ---------------- Kernel 2: per-expert capacity scan ------------------------
// 8 expert counters packed as 4x16-bit fields in two u64s (max 8192 < 65536).
__global__ __launch_bounds__(1024) void scan_kernel(const int* __restrict__ best,
                                                    const float* __restrict__ gate,
                                                    float* __restrict__ scale) {
    __shared__ unsigned long long s0[1024];
    __shared__ unsigned long long s1[1024];
    __shared__ unsigned long long tot[2];
    int t = threadIdx.x;
    unsigned long long base0 = 0, base1 = 0;

    for (int c = 0; c < NTOK / 1024; ++c) {
        int n = c * 1024 + t;
        int e = best[n];
        unsigned long long oh0 = (e < 4) ? (1ull << (16 * e)) : 0ull;
        unsigned long long oh1 = (e >= 4) ? (1ull << (16 * (e - 4))) : 0ull;
        unsigned long long v0 = oh0, v1 = oh1;
        s0[t] = v0; s1[t] = v1;
        __syncthreads();
        for (int off = 1; off < 1024; off <<= 1) {
            unsigned long long a0 = 0, a1 = 0;
            if (t >= off) { a0 = s0[t - off]; a1 = s1[t - off]; }
            __syncthreads();
            v0 += a0; v1 += a1;
            s0[t] = v0; s1[t] = v1;
            __syncthreads();
        }
        unsigned long long ex0 = base0 + (v0 - oh0);
        unsigned long long ex1 = base1 + (v1 - oh1);
        unsigned rank = (e < 4) ? (unsigned)((ex0 >> (16 * e)) & 0xFFFFull)
                                : (unsigned)((ex1 >> (16 * (e - 4))) & 0xFFFFull);
        scale[n] = (rank < CAP) ? gate[n] : 0.0f;
        if (t == 1023) { tot[0] = v0; tot[1] = v1; }
        __syncthreads();
        base0 += tot[0]; base1 += tot[1];
        __syncthreads();
    }
}

// ---------------- Kernel 3: out = scale[m] * (h @ We + be) ------------------
// bf16x3 split-precision GEMM on V_WMMA_F32_16X16X32_BF16.
// 256 threads = 8 waves (4x2), block tile 128x64, 32x32 per wave.
#define MT 128
#define NT 64
#define KC 32
#define APITCH 40   // ushorts per A row  (32 data + 8 pad) = 80 B
#define BPITCH 40   // ushorts per B col-row (transposed)   = 80 B

__global__ __launch_bounds__(256) void moe_gemm_kernel(const float* __restrict__ h,
                                                       const float* __restrict__ We,
                                                       const float* __restrict__ be,
                                                       const float* __restrict__ scale,
                                                       float* __restrict__ out) {
    __shared__ __align__(16) unsigned short Ah[MT * APITCH];  // 10 KB
    __shared__ __align__(16) unsigned short Al[MT * APITCH];  // 10 KB
    __shared__ __align__(16) unsigned short Bh[NT * BPITCH];  //  5 KB (col-major)
    __shared__ __align__(16) unsigned short Bl[NT * BPITCH];  //  5 KB

    int tid  = threadIdx.x;
    int lane = tid & 31;
    int wave = tid >> 5;
    int wm = wave >> 1;          // 0..3
    int wn = wave & 1;           // 0..1
    int mlane = lane & 15;
    int hl    = lane >> 4;       // 0 or 1
    int m0 = blockIdx.y * MT;
    int n0 = blockIdx.x * NT;

    v8f acc00 = {}, acc01 = {}, acc10 = {}, acc11 = {};

    int ar  = tid >> 3;          // 0..31
    int ac4 = (tid & 7) * 4;     // f32 col 0..28
    int br  = tid >> 4;          // 0..15  (k row)
    int bc4 = (tid & 15) * 4;    // n col 0..60

    for (int k0 = 0; k0 < DIM; k0 += KC) {
        v4f a_reg[4];
        v4f b_reg[2];
#pragma unroll
        for (int p = 0; p < 4; ++p)
            a_reg[p] = *(const v4f*)(h + (size_t)(m0 + ar + 32 * p) * DIM + k0 + ac4);
#pragma unroll
        for (int p = 0; p < 2; ++p)
            b_reg[p] = *(const v4f*)(We + (size_t)(k0 + br + 16 * p) * DIM + n0 + bc4);
        __syncthreads();    // previous chunk's compute done

        // A planes, row-major (K-contiguous)
#pragma unroll
        for (int p = 0; p < 4; ++p) {
            int row = ar + 32 * p;
            v4f v = a_reg[p];
            uint2 h2, l2;
            h2.x = pack_hi2(v.x, v.y); h2.y = pack_hi2(v.z, v.w);
            l2.x = pack_lo2(v.x, v.y); l2.y = pack_lo2(v.z, v.w);
            *(uint2*)(Ah + row * APITCH + ac4) = h2;
            *(uint2*)(Al + row * APITCH + ac4) = l2;
        }
        // B planes, stored transposed: B[col][k] (K-contiguous per column)
#pragma unroll
        for (int p = 0; p < 2; ++p) {
            int kk = br + 16 * p;    // 0..31
            v4f v = b_reg[p];
#pragma unroll
            for (int i = 0; i < 4; ++i) {
                float w = (i == 0) ? v.x : (i == 1) ? v.y : (i == 2) ? v.z : v.w;
                int col = bc4 + i;
                Bh[col * BPITCH + kk] = (unsigned short)(__float_as_uint(w) >> 16);
                Bl[col * BPITCH + kk] = bf16b(w - hi_part(w));
            }
        }
        __syncthreads();

        // Fragments: A lanes<16 hold K 0-7 / 16-23, lanes>=16 hold 8-15 / 24-31.
        //            B lanes<16 hold K 0-7 / 16-23, lanes>=16 hold 8-15 / 24-31.
        fragU ah0, ah1, al0, al1, bh0, bh1, bl0, bl1;
        int arow0 = (wm * 32 + mlane) * APITCH + hl * 8;
        int arow1 = arow0 + 16 * APITCH;
        ah0.h[0] = *(const v8bf*)(Ah + arow0);      ah0.h[1] = *(const v8bf*)(Ah + arow0 + 16);
        al0.h[0] = *(const v8bf*)(Al + arow0);      al0.h[1] = *(const v8bf*)(Al + arow0 + 16);
        ah1.h[0] = *(const v8bf*)(Ah + arow1);      ah1.h[1] = *(const v8bf*)(Ah + arow1 + 16);
        al1.h[0] = *(const v8bf*)(Al + arow1);      al1.h[1] = *(const v8bf*)(Al + arow1 + 16);
        int bcol0 = (wn * 32 + mlane) * BPITCH + hl * 8;
        int bcol1 = bcol0 + 16 * BPITCH;
        bh0.h[0] = *(const v8bf*)(Bh + bcol0);      bh0.h[1] = *(const v8bf*)(Bh + bcol0 + 16);
        bl0.h[0] = *(const v8bf*)(Bl + bcol0);      bl0.h[1] = *(const v8bf*)(Bl + bcol0 + 16);
        bh1.h[0] = *(const v8bf*)(Bh + bcol1);      bh1.h[1] = *(const v8bf*)(Bh + bcol1 + 16);
        bl1.h[0] = *(const v8bf*)(Bl + bcol1);      bl1.h[1] = *(const v8bf*)(Bl + bcol1 + 16);

        // A*B ~= Ah*Bh + Al*Bh + Ah*Bl  (Al*Bl ~ 2^-16, dropped)
        acc00 = __builtin_amdgcn_wmma_f32_16x16x32_bf16(false, ah0.v, false, bh0.v, (short)0, acc00, false, false);
        acc01 = __builtin_amdgcn_wmma_f32_16x16x32_bf16(false, ah0.v, false, bh1.v, (short)0, acc01, false, false);
        acc10 = __builtin_amdgcn_wmma_f32_16x16x32_bf16(false, ah1.v, false, bh0.v, (short)0, acc10, false, false);
        acc11 = __builtin_amdgcn_wmma_f32_16x16x32_bf16(false, ah1.v, false, bh1.v, (short)0, acc11, false, false);
        acc00 = __builtin_amdgcn_wmma_f32_16x16x32_bf16(false, al0.v, false, bh0.v, (short)0, acc00, false, false);
        acc01 = __builtin_amdgcn_wmma_f32_16x16x32_bf16(false, al0.v, false, bh1.v, (short)0, acc01, false, false);
        acc10 = __builtin_amdgcn_wmma_f32_16x16x32_bf16(false, al1.v, false, bh0.v, (short)0, acc10, false, false);
        acc11 = __builtin_amdgcn_wmma_f32_16x16x32_bf16(false, al1.v, false, bh1.v, (short)0, acc11, false, false);
        acc00 = __builtin_amdgcn_wmma_f32_16x16x32_bf16(false, ah0.v, false, bl0.v, (short)0, acc00, false, false);
        acc01 = __builtin_amdgcn_wmma_f32_16x16x32_bf16(false, ah0.v, false, bl1.v, (short)0, acc01, false, false);
        acc10 = __builtin_amdgcn_wmma_f32_16x16x32_bf16(false, ah1.v, false, bl0.v, (short)0, acc10, false, false);
        acc11 = __builtin_amdgcn_wmma_f32_16x16x32_bf16(false, ah1.v, false, bl1.v, (short)0, acc11, false, false);
    }

    // Epilogue: out[m][n] = scale[m] * (acc + be[n]); dropped tokens scale 0.
    int ncol0 = n0 + wn * 32 + mlane;
    float bias0 = be[ncol0];
    float bias1 = be[ncol0 + 16];
#pragma unroll
    for (int tm = 0; tm < 2; ++tm) {
        v8f accL = (tm == 0) ? acc00 : acc10;
        v8f accR = (tm == 0) ? acc01 : acc11;
#pragma unroll
        for (int j = 0; j < 8; ++j) {
            int m = m0 + wm * 32 + tm * 16 + j + 8 * hl;
            float s = scale[m];
            out[(size_t)m * DIM + ncol0]      = s * (accL[j] + bias0);
            out[(size_t)m * DIM + ncol0 + 16] = s * (accR[j] + bias1);
        }
    }
}

extern "C" void kernel_launch(void* const* d_in, const int* in_sizes, int n_in,
                              void* d_out, int out_size, void* d_ws, size_t ws_size,
                              hipStream_t stream) {
    const float* x  = (const float*)d_in[0];   // [B,S,D] f32
    const float* Wg = (const float*)d_in[1];   // [D,E]
    const float* We = (const float*)d_in[2];   // [D,D]
    const float* be = (const float*)d_in[3];   // [D]
    float* out = (float*)d_out;

    float* gate  = (float*)d_ws;
    int*   best  = (int*)((char*)d_ws + (size_t)NTOK * sizeof(float));
    float* scale = (float*)((char*)d_ws + 2 * (size_t)NTOK * sizeof(float));

    router_kernel<<<NTOK / 8, 256, 0, stream>>>(x, Wg, gate, best);
    scan_kernel<<<1, 1024, 0, stream>>>(best, gate, scale);
    dim3 grid(DIM / NT, NTOK / MT);
    moe_gemm_kernel<<<grid, 256, 0, stream>>>(x, We, be, scale, out);
}
